// MultiHeadAttention_77438260347155
// MI455X (gfx1250) — matrix operand order repaired and checked by reading, not compiled, not run on hardware
//
#include <hip/hip_runtime.h>

#define DEV __device__ __forceinline__

typedef __bf16 bf16;
typedef __attribute__((ext_vector_type(8)))  bf16  v8bf;
typedef __attribute__((ext_vector_type(16))) bf16  v16bf;
typedef __attribute__((ext_vector_type(8)))  float v8f;

constexpr int B = 4, T = 2048, D = 1024, H = 16, HD = 64;
constexpr float LOG2E = 1.4426950408889634f;

DEV bf16 f2bf(float f) {
  unsigned u = __builtin_bit_cast(unsigned, f);
  u += 0x7FFFu + ((u >> 16) & 1u);               // round-to-nearest-even
  unsigned short h = (unsigned short)(u >> 16);
  return __builtin_bit_cast(bf16, h);
}

DEV v16bf cat16(v8bf lo, v8bf hi) {
  return __builtin_shufflevector(lo, hi, 0,1,2,3,4,5,6,7,8,9,10,11,12,13,14,15);
}

DEV v8f wmma_bf16(v16bf a, v16bf b, v8f c) {
  return __builtin_amdgcn_wmma_f32_16x16x32_bf16(false, a, false, b, (short)0, c,
                                                 false, false);
}

DEV float rmax16(float v) {
  #pragma unroll
  for (int m = 1; m <= 8; m <<= 1) v = fmaxf(v, __shfl_xor(v, m, 32));
  return v;
}
DEV float rsum16(float v) {
  #pragma unroll
  for (int m = 1; m <= 8; m <<= 1) v += __shfl_xor(v, m, 32);
  return v;
}

// ---------------- prep kernels -------------------------------------------
__global__ void k_cvt_bf16(const float* __restrict__ in, bf16* __restrict__ out, int n) {
  int i = blockIdx.x * blockDim.x + threadIdx.x;
  if (i < n) out[i] = f2bf(in[i]);
}

// [H][D][HD] fp32 -> [H][HD][D] bf16
__global__ void k_tr_head(const float* __restrict__ in, bf16* __restrict__ out) {
  int idx = blockIdx.x * blockDim.x + threadIdx.x;   // over H*HD*D
  int d = idx % D;
  int rem = idx / D;
  int e = rem % HD;
  int h = rem / HD;
  out[idx] = f2bf(in[((size_t)h * D + d) * HD + e]);
}

// [D][D] fp32 -> transpose [n][d] bf16
__global__ void k_tr_o(const float* __restrict__ in, bf16* __restrict__ out) {
  int idx = blockIdx.x * blockDim.x + threadIdx.x;   // over D*D
  int d = idx % D;
  int n = idx / D;
  out[idx] = f2bf(in[(size_t)d * D + n]);
}

// ---------------- fused QKV projection -----------------------------------
// grid (T/16, B, H), block 128 (4 waves; wave w -> output cols [16w,16w+16))
// Register double-buffered over the K dimension so loads for step kk+32 are
// in flight while the three WMMAs for step kk execute.
__global__ __launch_bounds__(128) void k_qkv(
    const bf16* __restrict__ Xb,
    const bf16* __restrict__ Wqt, const bf16* __restrict__ Wkt,
    const bf16* __restrict__ Wvt,
    const float* __restrict__ bqf, const float* __restrict__ bkf,
    const float* __restrict__ bvf,
    bf16* __restrict__ Q, bf16* __restrict__ Kq, bf16* __restrict__ Vt) {
  const int lane = threadIdx.x & 31;
  const int w    = threadIdx.x >> 5;
  const int hi   = lane >> 4;
  const int lo16 = lane & 15;
  const int t0 = blockIdx.x * 16, b = blockIdx.y, h = blockIdx.z;

  const bf16* xrow = Xb + ((size_t)b * T + t0 + lo16) * (size_t)D;
  const int ecol = w * 16 + lo16;                       // N column
  const size_t wro = ((size_t)h * HD + ecol) * (size_t)D + 16 * hi;
  const bf16* wq = Wqt + wro;
  const bf16* wk = Wkt + wro;
  const bf16* wv = Wvt + wro;

  v8f aq = {}, ak = {}, av = {};
  // prologue: load K-step 0
  v16bf a  = cat16(*(const v8bf*)(xrow + 16 * hi), *(const v8bf*)(xrow + 8 + 16 * hi));
  v16bf mq = *(const v16bf*)(wq);
  v16bf mk = *(const v16bf*)(wk);
  v16bf mv = *(const v16bf*)(wv);
  #pragma unroll 4
  for (int kk = 0; kk < D; kk += 32) {
    const int kn = (kk + 32) & (D - 1);     // next step (wraps in-bounds)
    v16bf a2  = cat16(*(const v8bf*)(xrow + kn + 16 * hi),
                      *(const v8bf*)(xrow + kn + 8 + 16 * hi));
    v16bf mq2 = *(const v16bf*)(wq + kn);
    v16bf mk2 = *(const v16bf*)(wk + kn);
    v16bf mv2 = *(const v16bf*)(wv + kn);
    aq = wmma_bf16(a, mq, aq);
    ak = wmma_bf16(a, mk, ak);
    av = wmma_bf16(a, mv, av);
    a = a2; mq = mq2; mk = mk2; mv = mv2;
  }

  const size_t bh = (size_t)b * H + h;
  const float vbq = bqf[h * HD + ecol];
  const float vbk = bkf[h * HD + ecol];
  const float vbv = bvf[h * HD + ecol];
  bf16* qp = Q  + (bh * T + t0) * (size_t)HD + ecol;
  bf16* kp = Kq + (bh * T + t0) * (size_t)HD + ecol;
  v8bf vpack;
  #pragma unroll
  for (int r = 0; r < 8; ++r) {
    int t = r + 8 * hi;
    // fold 1/sqrt(HD) AND log2(e) into Q -> softmax runs in exp2 domain
    qp[(size_t)t * HD] = f2bf((aq[r] + vbq) * (0.125f * LOG2E));
    kp[(size_t)t * HD] = f2bf(ak[r] + vbk);
    vpack[r] = f2bf(av[r] + vbv);
  }
  // V transposed [e][t]: r runs over consecutive t -> one 16B store
  *(v8bf*)(Vt + (bh * HD + ecol) * (size_t)T + t0 + 8 * hi) = vpack;
}

// ---------------- flash attention ----------------------------------------
// grid (T/64, B, H), block 128; each wave owns one 16-query tile
__global__ __launch_bounds__(128) void k_attn(
    const bf16* __restrict__ Q, const bf16* __restrict__ Kq,
    const bf16* __restrict__ Vt, bf16* __restrict__ Cat) {
  __shared__ __align__(16) bf16 plds[4][16 * 32];
  const int lane = threadIdx.x & 31;
  const int warp = threadIdx.x >> 5;
  const int hi = lane >> 4, lo16 = lane & 15;
  const int b = blockIdx.y, h = blockIdx.z;
  const int t0 = (blockIdx.x * 4 + warp) * 16;
  const size_t bh = (size_t)b * H + h;

  const bf16* qrow = Q + (bh * T + t0 + lo16) * (size_t)HD;
  v16bf qa0 = cat16(*(const v8bf*)(qrow + 16 * hi),
                    *(const v8bf*)(qrow + 8 + 16 * hi));
  v16bf qa1 = cat16(*(const v8bf*)(qrow + 32 + 16 * hi),
                    *(const v8bf*)(qrow + 40 + 16 * hi));
  const bf16* kb = Kq + bh * (size_t)T * HD;
  const bf16* vb = Vt + bh * (size_t)HD * T;

  float m[8], l[8];
  v8f o0 = {}, o1 = {}, o2 = {}, o3 = {};
  #pragma unroll
  for (int r = 0; r < 8; ++r) { m[r] = -1e30f; l[r] = 0.f; }

  const int send = t0 + 16;                     // keys 0..t0+15 needed
  for (int sb = 0; sb < send; sb += 32) {       // 32-key blocks
    // S = Q*K^T  (16x32 = two C tiles, two K=32 steps over d)
    v8f st0 = {}, st1 = {};
    {
      const bf16* k0 = kb + (size_t)(sb + lo16) * HD + 16 * hi;
      const bf16* k1 = kb + (size_t)(sb + 16 + lo16) * HD + 16 * hi;
      st0 = wmma_bf16(qa0, *(const v16bf*)(k0), st0);
      st1 = wmma_bf16(qa0, *(const v16bf*)(k1), st1);
      st0 = wmma_bf16(qa1, *(const v16bf*)(k0 + 32), st0);
      st1 = wmma_bf16(qa1, *(const v16bf*)(k1 + 32), st1);
    }
    // issue V B-tile loads now; they overlap the softmax VALU work below
    const bf16* vrow = vb + sb + 16 * hi;
    v16bf vm0 = *(const v16bf*)(vrow + (size_t)(lo16) * T);
    v16bf vm1 = *(const v16bf*)(vrow + (size_t)(16 + lo16) * T);
    v16bf vm2 = *(const v16bf*)(vrow + (size_t)(32 + lo16) * T);
    v16bf vm3 = *(const v16bf*)(vrow + (size_t)(48 + lo16) * T);

    // causal mask (only diagonal blocks hit this; condition is wave-uniform)
    if (sb + 31 > t0) {
      #pragma unroll
      for (int r = 0; r < 8; ++r) {
        int row = t0 + r + 8 * hi;
        if (sb + lo16      > row) st0[r] = -1e30f;
        if (sb + 16 + lo16 > row) st1[r] = -1e30f;
      }
    }
    // online softmax (exp2 domain; log2e pre-folded into Q)
    float mn[8], sc[8];
    #pragma unroll
    for (int r = 0; r < 8; ++r) {
      float v = rmax16(fmaxf(st0[r], st1[r]));
      mn[r] = fmaxf(m[r], v);
      sc[r] = __builtin_amdgcn_exp2f(m[r] - mn[r]);
      m[r]  = mn[r];
      l[r] *= sc[r];
    }
    #pragma unroll
    for (int r = 0; r < 8; ++r) {
      o0[r] *= sc[r]; o1[r] *= sc[r]; o2[r] *= sc[r]; o3[r] *= sc[r];
    }
    // P = exp2(S-m): accumulate row sums, stage row-major bf16 in LDS
    #pragma unroll
    for (int r = 0; r < 8; ++r) {
      float p0 = __builtin_amdgcn_exp2f(st0[r] - mn[r]);
      float p1 = __builtin_amdgcn_exp2f(st1[r] - mn[r]);
      l[r] += rsum16(p0 + p1);
      int row = r + 8 * hi;
      plds[warp][row * 32 + lo16]      = f2bf(p0);
      plds[warp][row * 32 + 16 + lo16] = f2bf(p1);
    }
    // reload P in WMMA A-layout (two contiguous 16B ds reads per lane)
    const v8bf* pr = (const v8bf*)plds[warp];
    v16bf pa = cat16(pr[lo16 * 4 + 2 * hi], pr[lo16 * 4 + 2 * hi + 1]);
    // O += P * V   (V transposed [e][t] -> contiguous B-loads, issued above)
    o0 = wmma_bf16(pa, vm0, o0);
    o1 = wmma_bf16(pa, vm1, o1);
    o2 = wmma_bf16(pa, vm2, o2);
    o3 = wmma_bf16(pa, vm3, o3);
  }

  // normalize and store concatenated context [b][t][h*HD+e] as bf16
  bf16* cp = Cat + ((size_t)b * T + t0) * (size_t)D + h * HD;
  #pragma unroll
  for (int r = 0; r < 8; ++r) {
    float inv = 1.0f / l[r];
    bf16* row = cp + (size_t)(r + 8 * hi) * D;
    row[lo16]      = f2bf(o0[r] * inv);
    row[16 + lo16] = f2bf(o1[r] * inv);
    row[32 + lo16] = f2bf(o2[r] * inv);
    row[48 + lo16] = f2bf(o3[r] * inv);
  }
}

// ---------------- output projection --------------------------------------
// grid (B*T/16, D/64), block 128 (4 waves; wave w -> 16 output cols)
// Register double-buffered like k_qkv.
__global__ __launch_bounds__(128) void k_oproj(
    const bf16* __restrict__ Cat, const bf16* __restrict__ Wot,
    const float* __restrict__ bo, float* __restrict__ out) {
  const int lane = threadIdx.x & 31, w = threadIdx.x >> 5;
  const int hi = lane >> 4, lo16 = lane & 15;
  const size_t mt = blockIdx.x;
  const int n0 = blockIdx.y * 64 + w * 16;
  const bf16* crow = Cat + (mt * 16 + lo16) * (size_t)D;
  const bf16* wrow = Wot + (size_t)(n0 + lo16) * D + 16 * hi;
  v8f acc = {};
  v16bf a  = cat16(*(const v8bf*)(crow + 16 * hi), *(const v8bf*)(crow + 8 + 16 * hi));
  v16bf bm = *(const v16bf*)(wrow);
  #pragma unroll 4
  for (int kk = 0; kk < D; kk += 32) {
    const int kn = (kk + 32) & (D - 1);
    v16bf a2  = cat16(*(const v8bf*)(crow + kn + 16 * hi),
                      *(const v8bf*)(crow + kn + 8 + 16 * hi));
    v16bf bm2 = *(const v16bf*)(wrow + kn);
    acc = wmma_bf16(a, bm, acc);
    a = a2; bm = bm2;
  }
  const float bias = bo[n0 + lo16];
  float* op = out + mt * 16 * (size_t)D + n0 + lo16;
  #pragma unroll
  for (int r = 0; r < 8; ++r)
    op[(size_t)(r + 8 * hi) * D] = acc[r] + bias;
}

// ---------------- launch --------------------------------------------------
extern "C" void kernel_launch(void* const* d_in, const int* in_sizes, int n_in,
                              void* d_out, int out_size, void* d_ws, size_t ws_size,
                              hipStream_t stream) {
  const float* X  = (const float*)d_in[0];
  const float* Wq = (const float*)d_in[1];
  const float* bq = (const float*)d_in[2];
  const float* Wk = (const float*)d_in[3];
  const float* bk = (const float*)d_in[4];
  const float* Wv = (const float*)d_in[5];
  const float* bv = (const float*)d_in[6];
  const float* Wo = (const float*)d_in[7];
  const float* bo = (const float*)d_in[8];
  float* out = (float*)d_out;

  char* p = (char*)d_ws;
  bf16* Xb  = (bf16*)p; p += (size_t)B * T * D * 2;
  bf16* Wqt = (bf16*)p; p += (size_t)H * HD * D * 2;
  bf16* Wkt = (bf16*)p; p += (size_t)H * HD * D * 2;
  bf16* Wvt = (bf16*)p; p += (size_t)H * HD * D * 2;
  bf16* Wot = (bf16*)p; p += (size_t)D * D * 2;
  bf16* Qb  = (bf16*)p; p += (size_t)B * H * T * HD * 2;
  bf16* Kb  = (bf16*)p; p += (size_t)B * H * T * HD * 2;
  bf16* Vt  = (bf16*)p; p += (size_t)B * H * T * HD * 2;
  bf16* Cat = (bf16*)p; p += (size_t)B * T * D * 2;

  const int nx = B * T * D;
  k_cvt_bf16<<<nx / 256, 256, 0, stream>>>(X, Xb, nx);
  const int nw = H * HD * D;
  k_tr_head<<<nw / 256, 256, 0, stream>>>(Wq, Wqt);
  k_tr_head<<<nw / 256, 256, 0, stream>>>(Wk, Wkt);
  k_tr_head<<<nw / 256, 256, 0, stream>>>(Wv, Wvt);
  k_tr_o<<<(D * D) / 256, 256, 0, stream>>>(Wo, Wot);

  k_qkv<<<dim3(T / 16, B, H), 128, 0, stream>>>(Xb, Wqt, Wkt, Wvt, bq, bk, bv,
                                                Qb, Kb, Vt);
  k_attn<<<dim3(T / 64, B, H), 128, 0, stream>>>(Qb, Kb, Vt, Cat);
  k_oproj<<<dim3(B * T / 16, D / 64), 128, 0, stream>>>(Cat, Wot, bo, out);
}
